// VisionParser_15324443312346
// MI455X (gfx1250) — matrix operand order repaired and checked by reading, not compiled, not run on hardware
//
#include <hip/hip_runtime.h>
#include <hip/hip_bf16.h>

// ---------------------------------------------------------------------------
// VisionParser: pixel->prototype soft assignment
//   x  = feature_maps [16,512,64,64] fp32 (channel-major)
//   p  = prototypes   [1,1024,512]   fp32
//   d2 = |x|^2 + |p|^2 - 2 x.p^T          (GEMM, f16 WMMA / f32 acc)
//   sims = softmax(-10*sqrt(max(d2,0)))   over K=1024
//   out  = sims as [16,1024,64,64] ++ [mask scalar]
//
// Round-4 fix: rounds 2/3 spilled the accumulators to scratch because the
// epilogue indexed acc[t][u] with a runtime `u` (dynamic array index =>
// SROA demotes the whole array to scratch, poisoning the WMMA loop too).
// Accumulators are now two separate constant-indexed arrays and the
// epilogue is an inlined lambda instantiated per row-tile.
// ---------------------------------------------------------------------------

typedef __attribute__((ext_vector_type(16))) _Float16 v16h;
typedef __attribute__((ext_vector_type(8)))  _Float16 h8;
typedef __attribute__((ext_vector_type(4)))  _Float16 h4;
typedef __attribute__((ext_vector_type(8)))  float    v8f;

#if defined(__has_builtin) && __has_builtin(__builtin_amdgcn_sched_barrier)
#define SCHED_FENCE() __builtin_amdgcn_sched_barrier(0)
#else
#define SCHED_FENCE()
#endif

#define C_DIM 512
#define K_DIM 1024
#define HW    4096
#define NBAT  16
#define MROWS 32   // pixel rows per workgroup
#define NSTEP (C_DIM / 32)

// ---------------------------------------------------------------------------
// Prep: prototypes fp32 -> f16 (row-major [K,C]), |p|^2 per proto, zero counts
// ---------------------------------------------------------------------------
__global__ __launch_bounds__(256) void vp_prep(const float* __restrict__ p,
                                               _Float16* __restrict__ ph,
                                               float* __restrict__ pnorm,
                                               int* __restrict__ counts) {
  const int k = blockIdx.x;       // 1024 blocks
  const int t = threadIdx.x;      // 256 threads
  float s = 0.f;
  for (int c = t; c < C_DIM; c += 256) {
    float v = p[(size_t)k * C_DIM + c];
    ph[(size_t)k * C_DIM + c] = (_Float16)v;
    s += v * v;
  }
  __shared__ float red[256];
  red[t] = s;
  __syncthreads();
  for (int off = 128; off > 0; off >>= 1) {
    if (t < off) red[t] += red[t + off];
    __syncthreads();
  }
  if (t == 0) pnorm[k] = red[0];
  if (t < NBAT) counts[k * NBAT + t] = 0;   // zero all 16*1024 counts
}

// ---------------------------------------------------------------------------
// Main: per WG = 32 pixel rows x full K=1024.  8 waves; each wave owns 128
// columns x 2 row-tiles = 16 WMMA 16x16 f32 accumulator tiles.  C reduced in
// 16 steps of 32 with v_wmma_f32_16x16x32_f16; each B fragment used twice.
// ---------------------------------------------------------------------------
__global__ __launch_bounds__(256, 1) void vp_main(const float* __restrict__ fm,
                                                  const _Float16* __restrict__ ph,
                                                  const float* __restrict__ pnorm,
                                                  float* __restrict__ out,
                                                  int* __restrict__ counts) {
  __shared__ _Float16 As[2][MROWS][40];  // ping-pong: 32 pix x 32 ch f16 (16B rows)
  __shared__ float    xnormS[MROWS];
  __shared__ float    wmaxS[8][16];
  __shared__ int      widxS[8][16];
  __shared__ float    wsumS[8][16];

  const int tid  = threadIdx.x;
  const int wave = tid >> 5;
  const int lane = tid & 31;
  const int g    = lane >> 4;   // lane group (WMMA half)
  const int n    = lane & 15;   // column-in-tile / row-in-A

  const int r0   = blockIdx.x * MROWS;  // first pixel-row of this WG
  const int b    = r0 >> 12;            // batch (4096 pixels per image)
  const int pix0 = r0 & (HW - 1);

  const float* fmb = fm + (size_t)b * C_DIM * HW + pix0;

  if (tid < MROWS) xnormS[tid] = 0.f;

  // staging: each wave stages 4 channels x 32 pixels per pass
  const int apix = lane;        // pixel this thread stages
  const int acg  = wave;        // channel group: c = acg*4 + i

  v8f acc0[8] = {};             // row-tile 0 (rows 0..15), col-tiles 0..7
  v8f acc1[8] = {};             // row-tile 1 (rows 16..31)
  float xsq = 0.f;

  // B base for this lane: proto row (wave*128 + n), K-chunk g*16
  const _Float16* pbase = ph + (size_t)(wave * 128 + n) * C_DIM + g * 16;

  // ---- prologue: stage step 0 into buffer 0
  {
    h4 hv;
#pragma unroll
    for (int i = 0; i < 4; ++i) {
      const float xv = fmb[(size_t)(acg * 4 + i) * HW + apix];
      xsq += xv * xv;
      hv[i] = (_Float16)xv;
    }
    *(h4*)&As[0][apix][acg * 4] = hv;
  }

  for (int s = 0; s < NSTEP; ++s) {
    __syncthreads();   // buf[s&1] ready; buf[(s+1)&1] free (last read in step s-1)

    // ---- issue global loads for step s+1 now; consume them after the WMMAs
    float xv[4];
    if (s + 1 < NSTEP) {
      const int c0n = (s + 1) * 32;
#pragma unroll
      for (int i = 0; i < 4; ++i)
        xv[i] = fmb[(size_t)(c0n + acg * 4 + i) * HW + apix];
    }

    // ---- A fragments (contiguous k order, not the ISA table order):
    // lane (g,n): elems 0..15 <- K=g*16+0..15 (matches contiguous B)
    const int cb = s & 1;
    union { v16h v; h8 h[2]; } a0, a1;
    a0.h[0] = *(const h8*)&As[cb][n][g * 16];
    a0.h[1] = *(const h8*)&As[cb][n][g * 16 + 8];
    a1.h[0] = *(const h8*)&As[cb][16 + n][g * 16];
    a1.h[1] = *(const h8*)&As[cb][16 + n][g * 16 + 8];

    // ---- 8 col-tiles, 1-deep B prefetch, <=2 B fragments live
    const _Float16* pb = pbase + s * 32;
    v16h bcur = *(const v16h*)(pb);
#pragma unroll
    for (int t = 0; t < 8; ++t) {
      v16h bnext = bcur;
      if (t < 7) bnext = *(const v16h*)(pb + (size_t)(t + 1) * 16 * C_DIM);
      acc0[t] = __builtin_amdgcn_wmma_f32_16x16x32_f16(
          false, a0.v, false, bcur, (short)0, acc0[t], false, false);
      acc1[t] = __builtin_amdgcn_wmma_f32_16x16x32_f16(
          false, a1.v, false, bcur, (short)0, acc1[t], false, false);
      SCHED_FENCE();   // keep the scheduler from hoisting all B loads
      bcur = bnext;
    }

    // ---- convert + store the prefetched x tile (waits on the global loads)
    if (s + 1 < NSTEP) {
      h4 hv;
#pragma unroll
      for (int i = 0; i < 4; ++i) {
        xsq += xv[i] * xv[i];
        hv[i] = (_Float16)xv[i];
      }
      *(h4*)&As[(s + 1) & 1][apix][acg * 4] = hv;
    }
  }

  atomicAdd(&xnormS[apix], xsq);   // 8 partial |x|^2 per pixel
  __syncthreads();

  float* outb = out + (size_t)b * K_DIM * HW + pix0;

  // ---- epilogue for one 16-row tile; all acc indexing is constant
  auto epilogue = [&](v8f (&accu)[8], const int u) {
    float xn[8];
#pragma unroll
    for (int r = 0; r < 8; ++r) xn[r] = xnormS[u * 16 + 8 * g + r];  // row M = 8g+r

    float rmax[8];
    int   ridx[8];
#pragma unroll
    for (int r = 0; r < 8; ++r) { rmax[r] = -3.4e38f; ridx[r] = 0x7fffffff; }

    // acc -> logits in place; per-row (max, argmax) over this wave's 128 cols
#pragma unroll
    for (int t = 0; t < 8; ++t) {
      const int k = wave * 128 + t * 16 + n;
      const float pn = pnorm[k];
#pragma unroll
      for (int r = 0; r < 8; ++r) {
        float d2 = xn[r] + pn - 2.0f * accu[t][r];
        d2 = fmaxf(d2, 0.0f);
        const float logit = -10.0f * __builtin_sqrtf(d2);  // -dist/0.1
        accu[t][r] = logit;
        if (logit > rmax[r] || (logit == rmax[r] && k < ridx[r])) {
          rmax[r] = logit; ridx[r] = k;
        }
      }
    }
    // butterfly (max,idx) across the 16 lanes of each group
#pragma unroll
    for (int m = 1; m <= 8; m <<= 1) {
#pragma unroll
      for (int r = 0; r < 8; ++r) {
        const float om = __shfl_xor(rmax[r], m, 32);
        const int   oi = __shfl_xor(ridx[r], m, 32);
        if (om > rmax[r] || (om == rmax[r] && oi < ridx[r])) {
          rmax[r] = om; ridx[r] = oi;
        }
      }
    }
    if (n == 0) {
#pragma unroll
      for (int r = 0; r < 8; ++r) {
        wmaxS[wave][8 * g + r] = rmax[r];
        widxS[wave][8 * g + r] = ridx[r];
      }
    }
    __syncthreads();

    // global (max, argmax) per row across the 8 waves
    float gmax[8]; int gidx[8];
#pragma unroll
    for (int r = 0; r < 8; ++r) {
      float mv = -3.4e38f; int id = 0x7fffffff;
#pragma unroll
      for (int w2 = 0; w2 < 8; ++w2) {
        const float v = wmaxS[w2][8 * g + r];
        const int  iv = widxS[w2][8 * g + r];
        if (v > mv || (v == mv && iv < id)) { mv = v; id = iv; }
      }
      gmax[r] = mv; gidx[r] = id;
    }

    // exp & row sums
    float rsum[8];
#pragma unroll
    for (int r = 0; r < 8; ++r) rsum[r] = 0.f;
#pragma unroll
    for (int t = 0; t < 8; ++t) {
#pragma unroll
      for (int r = 0; r < 8; ++r) {
        const float e = __expf(accu[t][r] - gmax[r]);
        accu[t][r] = e;
        rsum[r] += e;
      }
    }
#pragma unroll
    for (int m = 1; m <= 8; m <<= 1) {
#pragma unroll
      for (int r = 0; r < 8; ++r) rsum[r] += __shfl_xor(rsum[r], m, 32);
    }
    if (n == 0) {
#pragma unroll
      for (int r = 0; r < 8; ++r) wsumS[wave][8 * g + r] = rsum[r];
    }
    __syncthreads();

    float rinv[8];
#pragma unroll
    for (int r = 0; r < 8; ++r) {
      float sacc = 0.f;
#pragma unroll
      for (int w2 = 0; w2 < 8; ++w2) sacc += wsumS[w2][8 * g + r];
      rinv[r] = 1.0f / sacc;
    }

    // write sims to transposed [b, k, pix] layout; 2x16B stores per tile
#pragma unroll
    for (int t = 0; t < 8; ++t) {
      const int k = wave * 128 + t * 16 + n;
      float4 lo, hi;
      lo.x = accu[t][0] * rinv[0]; lo.y = accu[t][1] * rinv[1];
      lo.z = accu[t][2] * rinv[2]; lo.w = accu[t][3] * rinv[3];
      hi.x = accu[t][4] * rinv[4]; hi.y = accu[t][5] * rinv[5];
      hi.z = accu[t][6] * rinv[6]; hi.w = accu[t][7] * rinv[7];
      float* dst = outb + (size_t)k * HW + u * 16 + 8 * g;
      *(float4*)(dst)     = lo;
      *(float4*)(dst + 4) = hi;
    }

    // per-row argmax -> (b,k) counts (lane 0: rows u*16+0..7, lane 16: +8..15)
    if (wave == 0 && n == 0) {
#pragma unroll
      for (int r = 0; r < 8; ++r)
        atomicAdd(&counts[b * K_DIM + gidx[r]], 1);
    }
  };

  epilogue(acc0, 0);
  epilogue(acc1, 1);
}

// ---------------------------------------------------------------------------
// Finalize: mask = sum(counts > 3) / BATCH_SIZE
// ---------------------------------------------------------------------------
__global__ __launch_bounds__(256) void vp_fin(const int* __restrict__ counts,
                                              float* __restrict__ out_mask) {
  __shared__ int red[256];
  const int t = threadIdx.x;
  int s = 0;
  for (int i = t; i < NBAT * K_DIM; i += 256) s += (counts[i] > 3) ? 1 : 0;
  red[t] = s;
  __syncthreads();
  for (int off = 128; off > 0; off >>= 1) {
    if (t < off) red[t] += red[t + off];
    __syncthreads();
  }
  if (t == 0) *out_mask = (float)red[0] / (float)NBAT;
}

// ---------------------------------------------------------------------------
extern "C" void kernel_launch(void* const* d_in, const int* in_sizes, int n_in,
                              void* d_out, int out_size, void* d_ws, size_t ws_size,
                              hipStream_t stream) {
  const float* fm = (const float*)d_in[0];   // [16,512,64,64]
  const float* pr = (const float*)d_in[1];   // [1,1024,512]
  float* out = (float*)d_out;                // 16*1024*64*64 sims + 1 mask

  char* ws = (char*)d_ws;
  _Float16* ph   = (_Float16*)ws;                                      // 1 MB
  float*    pnrm = (float*)(ws + (size_t)K_DIM * C_DIM * 2);           // 4 KB
  int*      cnts = (int*)(ws + (size_t)K_DIM * C_DIM * 2 + K_DIM * 4); // 64 KB

  vp_prep<<<K_DIM, 256, 0, stream>>>(pr, ph, pnrm, cnts);
  vp_main<<<(NBAT * HW) / MROWS, 256, 0, stream>>>(fm, ph, pnrm, out, cnts);
  vp_fin<<<1, 256, 0, stream>>>(cnts, out + (size_t)NBAT * K_DIM * HW);
}
